// DualEncoderEpsNetwork_10565619548387
// MI455X (gfx1250) — compile-verified, matrix operand order checked by reading, not run on hardware
//
#include <hip/hip_runtime.h>
#include <math.h>

// ---------------------------------------------------------------------------
// DualEncoderEpsNetwork (GeoDiff/AGDIFF) forward for MI455X (gfx1250, wave32).
// All H x H GEMMs go through v_wmma_f32_16x16x32_f16 (f16 in, f32 accumulate).
// Template-specialized GEMM: no runtime branches in the WMMA inner loop.
// ---------------------------------------------------------------------------

#define HDIM 128
#define BM   128       // rows per workgroup tile
#define BKG  128       // K chunk staged in LDS
#define LOG2_CONST 0.6931471805599453f

typedef __attribute__((ext_vector_type(16))) _Float16 v16h;
typedef __attribute__((ext_vector_type(8)))  _Float16 v8h;
typedef __attribute__((ext_vector_type(8)))  float    v8f;

__device__ __forceinline__ float sspf(float x) {
  // shifted softplus: log(1+e^x) - log(2), numerically stable
  float sp = (x > 20.0f) ? x : log1pf(__expf(x));
  return sp - LOG2_CONST;
}

__device__ __forceinline__ v16h frag16(const _Float16* p) {
  // A/B fragment: two contiguous 8-half (16B) chunks, 16 halves apart
  v8h lo = *(const v8h*)(p);
  v8h hi = *(const v8h*)(p + 16);
  return __builtin_shufflevector(lo, hi, 0, 1, 2, 3, 4, 5, 6, 7,
                                          8, 9, 10, 11, 12, 13, 14, 15);
}

// ---------------------------------------------------------------------------
// WMMA GEMM: OUT(M, NT*16) = epi( A(M,K,f32) @ W + bias )
// W supplied transposed as WT(N,K) row-major f16.
// ACT: 0 none, 1 relu, 2 shifted-softplus.  RS: per-row scale.  RD: residual.
// Block: 256 threads = 8 waves; wave owns a 16-row strip x NT 16x16 tiles.
// ---------------------------------------------------------------------------
template <int NT, int ACT, bool RS, bool RD>
__global__ __launch_bounds__(256)
void gemm_wmma_kernel(const float* __restrict__ A,
                      const _Float16* __restrict__ WT,
                      const float* __restrict__ bias,
                      const float* __restrict__ rowscale,
                      const float* __restrict__ resid,
                      float* __restrict__ OUT,
                      int M, int K)
{
  constexpr int N = NT * 16;
  __shared__ _Float16 Alds[BM * BKG];   // 32 KB
  __shared__ _Float16 Wlds[N * BKG];    // 32 KB (NT=8) / 16 KB (NT=4)

  const int tid  = threadIdx.x;
  const int wave = tid >> 5;
  const int lane = tid & 31;
  const int row0 = blockIdx.x * BM;

  v8f acc[NT];
#pragma unroll
  for (int t = 0; t < NT; ++t)
#pragma unroll
    for (int j = 0; j < 8; ++j) acc[t][j] = 0.0f;

  const int m  = lane & 15;             // M-row (A frag) / N-row of WT (B frag)
  const int k0 = (lane >> 4) << 3;      // lanes 16-31 hold the +8 K chunk

  for (int kk = 0; kk < K; kk += BKG) {
    // ---- stage A tile (f32 global -> f16 LDS), 128x128 ----
    for (int l = tid * 4; l < BM * BKG; l += 256 * 4) {
      const int r = l >> 7;             // BKG == 128
      const int c = l & (BKG - 1);
      const int gr = row0 + r;
      float x0 = 0.f, x1 = 0.f, x2 = 0.f, x3 = 0.f;
      if (gr < M) {
        const float4 v = *(const float4*)(A + (size_t)gr * K + kk + c);
        x0 = v.x; x1 = v.y; x2 = v.z; x3 = v.w;
      }
      _Float16* p = Alds + r * BKG + c;
      p[0] = (_Float16)x0; p[1] = (_Float16)x1;
      p[2] = (_Float16)x2; p[3] = (_Float16)x3;
    }
    // ---- stage WT tile (f16 -> f16 LDS), N x 128 ----
    for (int l = tid * 8; l < N * BKG; l += 256 * 8) {
      const int r = l >> 7;
      const int c = l & (BKG - 1);
      *(v8h*)(Wlds + r * BKG + c) = *(const v8h*)(WT + (size_t)r * K + kk + c);
    }
    __syncthreads();

    // ---- compute: 4 K-steps of 32, NT column tiles, fully unrolled ----
#pragma unroll
    for (int k32 = 0; k32 < BKG; k32 += 32) {
      const v16h a = frag16(Alds + (wave * 16 + m) * BKG + k32 + k0);
#pragma unroll
      for (int ct = 0; ct < NT; ++ct) {
        const v16h b = frag16(Wlds + (ct * 16 + m) * BKG + k32 + k0);
        acc[ct] = __builtin_amdgcn_wmma_f32_16x16x32_f16(
            false, a, false, b, (short)0, acc[ct], false, false);
      }
    }
    __syncthreads();
  }

  // ---- epilogue: lane holds col (ct*16 + lane&15), rows (lane>>4)*8 + i ----
  const int mr0 = (lane >> 4) << 3;
#pragma unroll
  for (int ct = 0; ct < NT; ++ct) {
    const int col = ct * 16 + (lane & 15);
    const float bv = bias ? bias[col] : 0.0f;
#pragma unroll
    for (int i = 0; i < 8; ++i) {
      const int gr = row0 + wave * 16 + mr0 + i;
      if (gr < M) {
        float v = acc[ct][i] + bv;
        if (ACT == 1)      v = fmaxf(v, 0.0f);
        else if (ACT == 2) v = sspf(v);
        if (RS) v *= rowscale[gr];
        const size_t o = (size_t)gr * N + col;
        if (RD) v += resid[o];
        OUT[o] = v;
      }
    }
  }
}

// ---------------------------------------------------------------------------
// Elementwise / gather / scatter helper kernels (float4 vectorized; H=128)
// ---------------------------------------------------------------------------

// transpose-convert weight: W(din,dout) f32 -> WT(dout,din) f16
__global__ void wt_convert_kernel(const float* __restrict__ W,
                                  _Float16* __restrict__ WT, int din, int dout)
{
  const long long total = (long long)din * dout;
  for (long long i = (long long)blockIdx.x * blockDim.x + threadIdx.x;
       i < total; i += (long long)gridDim.x * blockDim.x) {
    const int o = (int)(i / din), k = (int)(i % din);
    WT[i] = (_Float16)W[(size_t)k * dout + o];
  }
}

// per-edge: distance, cutoff mask C, local mask
__global__ void edge_scalar_kernel(const float* __restrict__ pos,
                                   const int* __restrict__ src,
                                   const int* __restrict__ dst,
                                   const int* __restrict__ etype,
                                   float* __restrict__ d, float* __restrict__ cm,
                                   float* __restrict__ lm, int E)
{
  for (int e = blockIdx.x * blockDim.x + threadIdx.x; e < E;
       e += gridDim.x * blockDim.x) {
    const int s = src[e], t = dst[e];
    const float dx = pos[3 * s + 0] - pos[3 * t + 0];
    const float dy = pos[3 * s + 1] - pos[3 * t + 1];
    const float dz = pos[3 * s + 2] - pos[3 * t + 2];
    const float dd = sqrtf(dx * dx + dy * dy + dz * dz);
    d[e]  = dd;
    cm[e] = (dd <= 10.0f) ? 1.0f : 0.0f;
    lm[e] = (etype[e] > 0) ? 1.0f : 0.0f;
  }
}

// edge encoder layer 0: y = relu(d * W1 + b1); per-(edge, 4-col group)
__global__ void edge_layer0_kernel(const float* __restrict__ d,
                                   const float* __restrict__ W1,
                                   const float* __restrict__ b1,
                                   float4* __restrict__ y, long long EQ)
{
  for (long long i = (long long)blockIdx.x * blockDim.x + threadIdx.x;
       i < EQ; i += (long long)gridDim.x * blockDim.x) {
    const int e = (int)(i >> 5), q = (int)(i & 31);
    const float4 w = ((const float4*)W1)[q];
    const float4 b = ((const float4*)b1)[q];
    const float dd = d[e];
    y[i] = make_float4(fmaxf(dd * w.x + b.x, 0.f), fmaxf(dd * w.y + b.y, 0.f),
                       fmaxf(dd * w.z + b.z, 0.f), fmaxf(dd * w.w + b.w, 0.f));
  }
}

// edge_attr = d_emb * bond_emb[edge_type]
__global__ void bond_mul_kernel(const float4* __restrict__ demb,
                                const float4* __restrict__ bond,
                                const int* __restrict__ etype,
                                float4* __restrict__ ea, long long EQ)
{
  for (long long i = (long long)blockIdx.x * blockDim.x + threadIdx.x;
       i < EQ; i += (long long)gridDim.x * blockDim.x) {
    const int e = (int)(i >> 5), q = (int)(i & 31);
    const float4 a = demb[i];
    const float4 b = bond[(size_t)etype[e] * 32 + q];
    ea[i] = make_float4(a.x * b.x, a.y * b.y, a.z * b.z, a.w * b.w);
  }
}

__global__ void gather_emb_kernel(const float4* __restrict__ emb,
                                  const int* __restrict__ atype,
                                  float4* __restrict__ out, long long NQ)
{
  for (long long i = (long long)blockIdx.x * blockDim.x + threadIdx.x;
       i < NQ; i += (long long)gridDim.x * blockDim.x) {
    const int n = (int)(i >> 5), q = (int)(i & 31);
    out[i] = emb[(size_t)atype[n] * 32 + q];
  }
}

__global__ void zero_kernel(float4* __restrict__ p, long long n4) {
  const float4 z = make_float4(0.f, 0.f, 0.f, 0.f);
  for (long long i = (long long)blockIdx.x * blockDim.x + threadIdx.x;
       i < n4; i += (long long)gridDim.x * blockDim.x) p[i] = z;
}

__global__ void add_kernel(const float4* __restrict__ a,
                           const float4* __restrict__ b,
                           float4* __restrict__ c, long long n4) {
  for (long long i = (long long)blockIdx.x * blockDim.x + threadIdx.x;
       i < n4; i += (long long)gridDim.x * blockDim.x) {
    const float4 x = a[i], y = b[i];
    c[i] = make_float4(x.x + y.x, x.y + y.y, x.z + y.z, x.w + y.w);
  }
}

__global__ void copy_kernel(const float4* __restrict__ a, float4* __restrict__ c,
                            long long n4) {
  for (long long i = (long long)blockIdx.x * blockDim.x + threadIdx.x;
       i < n4; i += (long long)gridDim.x * blockDim.x) c[i] = a[i];
}

// SchNet CFConv message: agg[dst] += lin1(h)[src] * We[e]  (segment_sum)
__global__ void scatter_mul_add_kernel(const float4* __restrict__ nodeF,
                                       const float4* __restrict__ edgeW,
                                       const int* __restrict__ src,
                                       const int* __restrict__ dst,
                                       float* __restrict__ agg, long long EQ)
{
  for (long long i = (long long)blockIdx.x * blockDim.x + threadIdx.x;
       i < EQ; i += (long long)gridDim.x * blockDim.x) {
    const int e = (int)(i >> 5), q = (int)(i & 31);
    const float4 nv = nodeF[(size_t)src[e] * 32 + q];
    const float4 wv = edgeW[i];
    float* base = agg + (size_t)dst[e] * HDIM + q * 4;
    atomicAdd(base + 0, nv.x * wv.x);
    atomicAdd(base + 1, nv.y * wv.y);
    atomicAdd(base + 2, nv.z * wv.z);
    atomicAdd(base + 3, nv.w * wv.w);
  }
}

// GIN message: agg[dst] += relu(x[src] + edge_attr) * local_mask
__global__ void gin_msg_kernel(const float4* __restrict__ x,
                               const float4* __restrict__ ea,
                               const float* __restrict__ lm,
                               const int* __restrict__ src,
                               const int* __restrict__ dst,
                               float* __restrict__ agg, long long EQ)
{
  for (long long i = (long long)blockIdx.x * blockDim.x + threadIdx.x;
       i < EQ; i += (long long)gridDim.x * blockDim.x) {
    const int e = (int)(i >> 5), q = (int)(i & 31);
    const float4 xv = x[(size_t)src[e] * 32 + q];
    const float4 av = ea[i];
    const float s = lm[e];
    float* base = agg + (size_t)dst[e] * HDIM + q * 4;
    atomicAdd(base + 0, fmaxf(xv.x + av.x, 0.f) * s);
    atomicAdd(base + 1, fmaxf(xv.y + av.y, 0.f) * s);
    atomicAdd(base + 2, fmaxf(xv.z + av.z, 0.f) * s);
    atomicAdd(base + 3, fmaxf(xv.w + av.w, 0.f) * s);
  }
}

// pair feature: P[e] = [ node[src]*node[dst] (128) , edge_attr (128) ]
__global__ void pairfeat_kernel(const float4* __restrict__ node,
                                const float4* __restrict__ ea,
                                const int* __restrict__ src,
                                const int* __restrict__ dst,
                                float4* __restrict__ P, long long EQ)
{
  for (long long i = (long long)blockIdx.x * blockDim.x + threadIdx.x;
       i < EQ; i += (long long)gridDim.x * blockDim.x) {
    const int e = (int)(i >> 5), q = (int)(i & 31);
    const float4 a = node[(size_t)src[e] * 32 + q];
    const float4 b = node[(size_t)dst[e] * 32 + q];
    P[(size_t)e * 64 + q] = make_float4(a.x * b.x, a.y * b.y, a.z * b.z, a.w * b.w);
    P[(size_t)e * 64 + 32 + q] = ea[i];
  }
}

// final 64 -> 1 linear (optionally * local_mask)
__global__ void final_dot_kernel(const float* __restrict__ T,
                                 const float* __restrict__ W,
                                 const float* __restrict__ b,
                                 const float* __restrict__ lm,
                                 float* __restrict__ out, int E)
{
  for (int e = blockIdx.x * blockDim.x + threadIdx.x; e < E;
       e += gridDim.x * blockDim.x) {
    const float4* t4 = (const float4*)(T + (size_t)e * 64);
    const float4* w4 = (const float4*)W;
    float s = b[0];
#pragma unroll
    for (int q = 0; q < 16; ++q) {
      const float4 t = t4[q], w = w4[q];
      s += t.x * w.x + t.y * w.y + t.z * w.z + t.w * w.w;
    }
    if (lm) s *= lm[e];
    out[e] = s;
  }
}

// ---------------------------------------------------------------------------
// Host-side orchestration
// ---------------------------------------------------------------------------
static inline dim3 gsz(long long n) {
  long long blk = (n + 255) / 256;
  if (blk > 65535) blk = 65535;
  if (blk < 1) blk = 1;
  return dim3((unsigned)blk);
}

template <int NT>
static void launch_nt(const float* A, const _Float16* WT, const float* bias,
                      const float* rowscale, const float* resid, float* out,
                      int M, int K, int act, hipStream_t s) {
  const dim3 g((M + BM - 1) / BM);
  const bool rs = rowscale != nullptr, rd = resid != nullptr;
  auto L = [&](auto kern) {
    kern<<<g, 256, 0, s>>>(A, WT, bias, rowscale, resid, out, M, K);
  };
  switch (act) {
    case 0:
      rs ? (rd ? L(gemm_wmma_kernel<NT, 0, true, true>)
               : L(gemm_wmma_kernel<NT, 0, true, false>))
         : (rd ? L(gemm_wmma_kernel<NT, 0, false, true>)
               : L(gemm_wmma_kernel<NT, 0, false, false>));
      break;
    case 1:
      rs ? (rd ? L(gemm_wmma_kernel<NT, 1, true, true>)
               : L(gemm_wmma_kernel<NT, 1, true, false>))
         : (rd ? L(gemm_wmma_kernel<NT, 1, false, true>)
               : L(gemm_wmma_kernel<NT, 1, false, false>));
      break;
    default:
      rs ? (rd ? L(gemm_wmma_kernel<NT, 2, true, true>)
               : L(gemm_wmma_kernel<NT, 2, true, false>))
         : (rd ? L(gemm_wmma_kernel<NT, 2, false, true>)
               : L(gemm_wmma_kernel<NT, 2, false, false>));
      break;
  }
}

static void launch_gemm(const float* A, const _Float16* WT, const float* bias,
                        const float* rowscale, const float* resid, float* out,
                        int M, int K, int N, int act, hipStream_t s) {
  if (N == 64) launch_nt<4>(A, WT, bias, rowscale, resid, out, M, K, act, s);
  else         launch_nt<8>(A, WT, bias, rowscale, resid, out, M, K, act, s);
}

extern "C" void kernel_launch(void* const* d_in, const int* in_sizes, int n_in,
                              void* d_out, int out_size, void* d_ws, size_t ws_size,
                              hipStream_t stream)
{
  const int Nn = in_sizes[0];
  const int E  = in_sizes[3];
  const long long EH = (long long)E * HDIM;
  const long long NH = (long long)Nn * HDIM;
  const long long EQ = EH >> 2;   // E * 32 float4 groups
  const long long NQ = NH >> 2;

  const int*   atype = (const int*)d_in[0];
  const float* pos   = (const float*)d_in[1];
  const int*   ei    = (const int*)d_in[2];
  const int*   etype = (const int*)d_in[3];
  const int*   esrc  = ei;
  const int*   edst  = ei + E;

  auto PF = [&](int i) { return (const float*)d_in[i]; };

  // jax pytree flatten order (dict keys sorted; None bias of lin1 dropped)
  const int Bp = 4;
  const int I_BOND = Bp + 0;
  const int I_EE_W0 = Bp + 1, I_EE_B0 = Bp + 2, I_EE_W1 = Bp + 3, I_EE_B1 = Bp + 4;
  auto GINP = [&](int c, int j) { return Bp + 5 + 4 * c + j; };   // l1W,l1b,l2W,l2b
  const int I_GIN_EMB = Bp + 21;
  const int I_MG = Bp + 22;  // W0 b0 W1 b1 W2 b2
  const int I_ML = Bp + 28;
  auto SBP = [&](int b, int j) { return Bp + 34 + 9 * b + j; };   // linW,linb,lin1W,lin2W,lin2b,mlp1W,mlp1b,mlp2W,mlp2b
  const int I_SCH_EMB = Bp + 70;

  // bump allocator over workspace
  char* wsp = (char*)d_ws;
  size_t off = 0;
  auto alloc = [&](size_t bytes) -> void* {
    void* p = wsp + off;
    off = (off + bytes + 255) & ~(size_t)255;
    return p;
  };
  auto convW = [&](const float* W, int din, int dout) -> _Float16* {
    _Float16* wt = (_Float16*)alloc((size_t)din * dout * sizeof(_Float16));
    wt_convert_kernel<<<gsz((long long)din * dout), 256, 0, stream>>>(W, wt, din, dout);
    return wt;
  };

  // ---- convert all GEMM weights to transposed f16 ----
  _Float16* wt_ee1 = convW(PF(I_EE_W1), HDIM, HDIM);
  _Float16 *wt_lin[4], *wt_lin1[4], *wt_lin2[4], *wt_mlp1[4], *wt_mlp2[4];
  for (int b = 0; b < 4; ++b) {
    wt_lin[b]  = convW(PF(SBP(b, 0)), HDIM, HDIM);
    wt_lin1[b] = convW(PF(SBP(b, 2)), HDIM, HDIM);
    wt_lin2[b] = convW(PF(SBP(b, 3)), HDIM, HDIM);
    wt_mlp1[b] = convW(PF(SBP(b, 5)), HDIM, HDIM);
    wt_mlp2[b] = convW(PF(SBP(b, 7)), HDIM, HDIM);
  }
  _Float16 *wt_l1g[4], *wt_l2g[4];
  for (int c = 0; c < 4; ++c) {
    wt_l1g[c] = convW(PF(GINP(c, 0)), HDIM, HDIM);
    wt_l2g[c] = convW(PF(GINP(c, 2)), HDIM, HDIM);
  }
  _Float16* wt_g0 = convW(PF(I_MG + 0), 2 * HDIM, HDIM);
  _Float16* wt_g1 = convW(PF(I_MG + 2), HDIM, 64);
  _Float16* wt_m0 = convW(PF(I_ML + 0), 2 * HDIM, HDIM);
  _Float16* wt_m1 = convW(PF(I_ML + 2), HDIM, 64);

  // ---- activation buffers ----
  float* dscal = (float*)alloc((size_t)E * 4);
  float* cmask = (float*)alloc((size_t)E * 4);
  float* lmf   = (float*)alloc((size_t)E * 4);
  float* eb1   = (float*)alloc((size_t)EH * 4);          // edge scratch A
  float* eb2   = (float*)alloc((size_t)EH * 4);          // edge scratch B
  float* eattr = (float*)alloc((size_t)EH * 4);          // edge_attr (persistent)
  float* Pbuf  = (float*)alloc((size_t)E * 2 * HDIM * 4);// pair features (E,256)
  float* hA    = (float*)alloc((size_t)NH * 4);
  float* hB    = (float*)alloc((size_t)NH * 4);
  float* xA    = (float*)alloc((size_t)NH * 4);
  float* xB    = (float*)alloc((size_t)NH * 4);
  float* nt1   = (float*)alloc((size_t)NH * 4);
  float* nt2   = (float*)alloc((size_t)NH * 4);
  float* agg   = (float*)alloc((size_t)NH * 4);
  float* ginin = (float*)alloc((size_t)NH * 4);

  float* out_f = (float*)d_out;  // [eig(E) | eil(E) | d(E) | mask(E)]

  // ---- 1) edge scalars + edge encoder ----
  edge_scalar_kernel<<<gsz(E), 256, 0, stream>>>(pos, esrc, edst, etype,
                                                 dscal, cmask, lmf, E);
  edge_layer0_kernel<<<gsz(EQ), 256, 0, stream>>>(dscal, PF(I_EE_W0), PF(I_EE_B0),
                                                  (float4*)eb1, EQ);
  launch_gemm(eb1, wt_ee1, PF(I_EE_B1), nullptr, nullptr, eb2,
              E, HDIM, HDIM, /*act*/0, stream);                   // d_emb
  bond_mul_kernel<<<gsz(EQ), 256, 0, stream>>>((const float4*)eb2,
                                               (const float4*)PF(I_BOND), etype,
                                               (float4*)eattr, EQ);

  // ---- 2) SchNet global encoder ----
  gather_emb_kernel<<<gsz(NQ), 256, 0, stream>>>((const float4*)PF(I_SCH_EMB),
                                                 atype, (float4*)hA, NQ);
  for (int b = 0; b < 4; ++b) {
    launch_gemm(eattr, wt_mlp1[b], PF(SBP(b, 6)), nullptr, nullptr, eb1,
                E, HDIM, HDIM, /*ssp*/2, stream);
    launch_gemm(eb1, wt_mlp2[b], PF(SBP(b, 8)), cmask, nullptr, eb2,
                E, HDIM, HDIM, /*none*/0, stream);                // W_e * C
    launch_gemm(hA, wt_lin1[b], nullptr, nullptr, nullptr, nt1,
                Nn, HDIM, HDIM, 0, stream);                       // lin1(h)
    zero_kernel<<<gsz(NQ), 256, 0, stream>>>((float4*)agg, NQ);
    scatter_mul_add_kernel<<<gsz(EQ), 256, 0, stream>>>(
        (const float4*)nt1, (const float4*)eb2, esrc, edst, agg, EQ);
    launch_gemm(agg, wt_lin2[b], PF(SBP(b, 4)), nullptr, nullptr, nt2,
                Nn, HDIM, HDIM, /*ssp*/2, stream);                // ssp(lin2(agg))
    launch_gemm(nt2, wt_lin[b], PF(SBP(b, 1)), nullptr, hA, hB,
                Nn, HDIM, HDIM, 0, stream);                       // h += lin(.)
    float* t = hA; hA = hB; hB = t;
  }
  // node_global = hA

  // ---- 3) GIN local encoder ----
  gather_emb_kernel<<<gsz(NQ), 256, 0, stream>>>((const float4*)PF(I_GIN_EMB),
                                                 atype, (float4*)xA, NQ);
  for (int c = 0; c < 4; ++c) {
    zero_kernel<<<gsz(NQ), 256, 0, stream>>>((float4*)agg, NQ);
    gin_msg_kernel<<<gsz(EQ), 256, 0, stream>>>(
        (const float4*)xA, (const float4*)eattr, lmf, esrc, edst, agg, EQ);
    add_kernel<<<gsz(NQ), 256, 0, stream>>>((const float4*)agg, (const float4*)xA,
                                            (float4*)ginin, NQ);  // agg + x
    launch_gemm(ginin, wt_l1g[c], PF(GINP(c, 1)), nullptr, nullptr, nt1,
                Nn, HDIM, HDIM, /*relu*/1, stream);
    launch_gemm(nt1, wt_l2g[c], PF(GINP(c, 3)), nullptr, xA, xB,
                Nn, HDIM, HDIM, (c < 3) ? 1 : 0, stream);         // x = act(.)+x
    float* t = xA; xA = xB; xB = t;
  }
  // node_local = xA

  // ---- 4) global pair MLP: 2H -> H(relu) -> H/2(relu) -> 1 ----
  pairfeat_kernel<<<gsz(EQ), 256, 0, stream>>>(
      (const float4*)hA, (const float4*)eattr, esrc, edst, (float4*)Pbuf, EQ);
  launch_gemm(Pbuf, wt_g0, PF(I_MG + 1), nullptr, nullptr, eb1,
              E, 2 * HDIM, HDIM, 1, stream);
  launch_gemm(eb1, wt_g1, PF(I_MG + 3), nullptr, nullptr, eb2,
              E, HDIM, 64, 1, stream);
  final_dot_kernel<<<gsz(E), 256, 0, stream>>>(eb2, PF(I_MG + 4), PF(I_MG + 5),
                                               nullptr, out_f, E);

  // ---- 5) local pair MLP (masked by local_mask) ----
  pairfeat_kernel<<<gsz(EQ), 256, 0, stream>>>(
      (const float4*)xA, (const float4*)eattr, esrc, edst, (float4*)Pbuf, EQ);
  launch_gemm(Pbuf, wt_m0, PF(I_ML + 1), nullptr, nullptr, eb1,
              E, 2 * HDIM, HDIM, 1, stream);
  launch_gemm(eb1, wt_m1, PF(I_ML + 3), nullptr, nullptr, eb2,
              E, HDIM, 64, 1, stream);
  final_dot_kernel<<<gsz(E), 256, 0, stream>>>(eb2, PF(I_ML + 4), PF(I_ML + 5),
                                               lmf, out_f + E, E);

  // ---- 6) aux outputs: d and local_mask ----
  copy_kernel<<<gsz(E / 4), 256, 0, stream>>>((const float4*)dscal,
                                              (float4*)(out_f + 2 * (size_t)E), E / 4);
  copy_kernel<<<gsz(E / 4), 256, 0, stream>>>((const float4*)lmf,
                                              (float4*)(out_f + 3 * (size_t)E), E / 4);

  (void)n_in; (void)out_size; (void)ws_size;
}
